// Res5ROIHeads_2267742732668
// MI455X (gfx1250) — matrix operand order repaired and checked by reading, not compiled, not run on hardware
//
#include <hip/hip_runtime.h>
#include <hip/hip_bf16.h>

// ---------------------------------------------------------------------------
// Res5ROIHeads for MI455X (gfx1250, wave32).
//  - ROI-Align gather is L2-resident (features = 32MB < 192MB L2); dominant cost.
//  - Head GEMM (512x1024 @ 1024x96) runs on v_wmma_f32_16x16x32_f16, with the
//    A tile staged to LDS once per block via global_load_async_to_lds_b128
//    (ASYNCcnt path) and read back as ds_load_b128 fragments.
//  - Outputs: det (2,256,6) | matched_idxs (2,256) | match_labels (2,256) = 4096 f32.
// ---------------------------------------------------------------------------

typedef _Float16 v8h  __attribute__((ext_vector_type(8)));
typedef _Float16 v16h __attribute__((ext_vector_type(16)));
typedef float    v8f  __attribute__((ext_vector_type(8)));

#define FEAT_H 50
#define FEAT_W 80
#define FEAT_C 1024
#define NPROP  256
#define NGT    64
#define NPOOL  14
#define NSAMP  196          // 14*14
#define NCOLP  96           // 85 head columns padded to 6 WMMA N-tiles
#define SCALE_CLAMP 4.135166556742356f

// ------------------------- 1. IoU matcher ----------------------------------
__global__ __launch_bounds__(256) void matcher_kernel(
    const float* __restrict__ gt, const float* __restrict__ props,
    float* __restrict__ out) {
  const int n = blockIdx.x;
  const int p = threadIdx.x;
  const float* pp = props + ((size_t)n * NPROP + p) * 4;
  const float px1 = pp[0], py1 = pp[1], px2 = pp[2], py2 = pp[3];
  const float parea = (px2 - px1) * (py2 - py1);
  float best = -1.0f;
  int bestg = 0;
  for (int g = 0; g < NGT; ++g) {
    const float* q = gt + ((size_t)n * NGT + g) * 4;
    const float ix1 = fmaxf(px1, q[0]), iy1 = fmaxf(py1, q[1]);
    const float ix2 = fminf(px2, q[2]), iy2 = fminf(py2, q[3]);
    const float iw = fmaxf(ix2 - ix1, 0.0f), ih = fmaxf(iy2 - iy1, 0.0f);
    const float inter = iw * ih;
    const float garea = (q[2] - q[0]) * (q[3] - q[1]);
    const float iou = inter / fmaxf(garea + parea - inter, 1e-9f);
    if (iou > best) { best = iou; bestg = g; }   // strict > keeps first (argmax tie rule)
  }
  out[3072 + n * NPROP + p] = (float)bestg;                 // matched_idxs
  out[3584 + n * NPROP + p] = (best >= 0.5f) ? 1.0f : 0.0f; // match_labels
}

// ------------- 2. pack [W_cls | W_box | 0] transposed to f16 ---------------
// WT[n][k], n in [0,96), k in [0,1024): column-major weights so each WMMA lane
// loads two runs of 8 contiguous K halves for its B column.
__global__ __launch_bounds__(256) void prep_weights_kernel(
    const float* __restrict__ Wcls, const float* __restrict__ Wbox,
    _Float16* __restrict__ WT) {
  const int idx = blockIdx.x * 256 + threadIdx.x;     // 96*1024 total
  if (idx >= NCOLP * FEAT_C) return;
  const int n = idx >> 10;
  const int k = idx & 1023;
  float v = 0.0f;
  if (n < 81)      v = Wcls[k * 81 + n];
  else if (n < 85) v = Wbox[k * 4 + (n - 81)];
  WT[(size_t)n * FEAT_C + k] = (_Float16)v;
}

// -------------------- 3. ROI-Align + spatial mean --------------------------
// One block per (image,box); 196 sample points precomputed to LDS; each of the
// 256 threads owns 4 channels (float4 gathers, 784 b128 loads/thread, L2-hot).
__global__ __launch_bounds__(256) void roi_align_kernel(
    const float* __restrict__ feat, const float* __restrict__ props,
    _Float16* __restrict__ out16) {
  __shared__ int   sIdx[NSAMP][4];
  __shared__ float sW[NSAMP][4];
  const int img = blockIdx.x >> 8;
  const int t = threadIdx.x;
  const float* p = props + (size_t)blockIdx.x * 4;
  const float x1 = p[0] * 0.0625f, y1 = p[1] * 0.0625f;
  const float x2 = p[2] * 0.0625f, y2 = p[3] * 0.0625f;
  const float bw = (x2 - x1) * (1.0f / NPOOL);
  const float bh = (y2 - y1) * (1.0f / NPOOL);
  if (t < NSAMP) {
    const int py = t / NPOOL, px = t % NPOOL;
    float gy = y1 + ((float)py + 0.5f) * bh - 0.5f;
    float gx = x1 + ((float)px + 0.5f) * bw - 0.5f;
    gy = fminf(fmaxf(gy, 0.0f), (float)(FEAT_H - 1));
    gx = fminf(fmaxf(gx, 0.0f), (float)(FEAT_W - 1));
    const float y0f = floorf(gy), x0f = floorf(gx);
    const int y0 = (int)y0f, x0 = (int)x0f;
    const int y1i = min(y0 + 1, FEAT_H - 1);
    const int x1i = min(x0 + 1, FEAT_W - 1);
    const float ly = gy - y0f, lx = gx - x0f;
    const float hy = 1.0f - ly, hx = 1.0f - lx;
    sIdx[t][0] = (y0  * FEAT_W + x0 ) * FEAT_C;
    sIdx[t][1] = (y0  * FEAT_W + x1i) * FEAT_C;
    sIdx[t][2] = (y1i * FEAT_W + x0 ) * FEAT_C;
    sIdx[t][3] = (y1i * FEAT_W + x1i) * FEAT_C;
    sW[t][0] = hy * hx; sW[t][1] = hy * lx;
    sW[t][2] = ly * hx; sW[t][3] = ly * lx;
  }
  __syncthreads();
  const float* fb = feat + (size_t)img * (FEAT_H * FEAT_W * FEAT_C) + t * 4;
  float4 acc = make_float4(0.0f, 0.0f, 0.0f, 0.0f);
  for (int s = 0; s < NSAMP; ++s) {
    const int i0 = sIdx[s][0], i1 = sIdx[s][1], i2 = sIdx[s][2], i3 = sIdx[s][3];
    const float w0 = sW[s][0], w1 = sW[s][1], w2 = sW[s][2], w3 = sW[s][3];
    if (s + 1 < NSAMP) __builtin_prefetch(fb + sIdx[s + 1][0], 0, 1); // global_prefetch_b8
    const float4 v0 = *(const float4*)(fb + i0);
    const float4 v1 = *(const float4*)(fb + i1);
    const float4 v2 = *(const float4*)(fb + i2);
    const float4 v3 = *(const float4*)(fb + i3);
    acc.x += w0 * v0.x + w1 * v1.x + w2 * v2.x + w3 * v3.x;
    acc.y += w0 * v0.y + w1 * v1.y + w2 * v2.y + w3 * v3.y;
    acc.z += w0 * v0.z + w1 * v1.z + w2 * v2.z + w3 * v3.z;
    acc.w += w0 * v0.w + w1 * v1.w + w2 * v2.w + w3 * v3.w;
  }
  const float inv = 1.0f / (float)NSAMP;
  _Float16* o = out16 + (size_t)blockIdx.x * FEAT_C + t * 4;
  o[0] = (_Float16)(acc.x * inv);
  o[1] = (_Float16)(acc.y * inv);
  o[2] = (_Float16)(acc.z * inv);
  o[3] = (_Float16)(acc.w * inv);
}

// ------------------- 4. head GEMM via v_wmma_f32_16x16x32_f16 --------------
// 32 blocks (16-row M tiles) x 6 waves (16-col N tiles).
// Stage the shared 32KB A tile to LDS once per block with the gfx1250 async
// LDS-DMA path (global_load_async_to_lds_b128, ASYNCcnt), removing the 6x
// redundant global A reads; fragments then come from ds_load_b128.
// A-frag per ISA layout: lane<16 holds M=lane, K={k0..k0+7, k0+16..k0+23};
// lane>=16 shifts K by 8. B-frag: lane holds column N=lane&15, same K
// split as A (lane>=16 -> +8).
__global__ __launch_bounds__(192) void gemm_wmma_kernel(
    const _Float16* __restrict__ A, const _Float16* __restrict__ BT,
    const float* __restrict__ b_cls, const float* __restrict__ b_box,
    float* __restrict__ logits) {
  __shared__ __align__(16) _Float16 sA[16 * FEAT_C];   // 32 KB A tile
  const int lane  = threadIdx.x & 31;
  const int wave  = threadIdx.x >> 5;          // N-tile 0..5
  const int mbase = blockIdx.x * 16;

  // ---- async DMA: global A tile -> LDS (2048 x 16B chunks, 192 threads) ----
  {
    const char* gsrc = (const char*)(A + (size_t)mbase * FEAT_C);
    const unsigned int lbase = (unsigned int)(uintptr_t)(&sA[0]); // flat->LDS aperture truncation
    for (int i = threadIdx.x; i < 2048; i += 192) {
      const unsigned long long gaddr = (unsigned long long)(uintptr_t)(gsrc + i * 16);
      const unsigned int       laddr = lbase + (unsigned int)(i * 16);
      asm volatile("global_load_async_to_lds_b128 %0, %1, off"
                   :: "v"(laddr), "v"(gaddr) : "memory");
    }
    asm volatile("s_wait_asynccnt 0x0" ::: "memory");
  }
  __syncthreads();

  const int arow = lane & 15;
  const int aklo = (lane >> 4) * 8;
  const int bcol = wave * 16 + (lane & 15);
  const int bklo = (lane >> 4) * 8;
  const _Float16* sa   = sA + (size_t)arow * FEAT_C;
  const _Float16* bptr = BT + (size_t)bcol * FEAT_C;

  v8f acc = {};
  for (int k0 = 0; k0 < FEAT_C; k0 += 32) {
    const v8h a0 = *(const v8h*)(sa + k0 + aklo);        // ds_load_b128
    const v8h a1 = *(const v8h*)(sa + k0 + aklo + 16);   // ds_load_b128
    const v8h b0 = *(const v8h*)(bptr + k0 + bklo);      // global_load_b128
    const v8h b1 = *(const v8h*)(bptr + k0 + bklo + 16); // global_load_b128
    const v16h a = __builtin_shufflevector(a0, a1, 0,1,2,3,4,5,6,7,8,9,10,11,12,13,14,15);
    const v16h b = __builtin_shufflevector(b0, b1, 0,1,2,3,4,5,6,7,8,9,10,11,12,13,14,15);
    acc = __builtin_amdgcn_wmma_f32_16x16x32_f16(
        /*neg_a=*/false, a, /*neg_b=*/false, b,
        /*c_mod=*/(short)0, acc, /*reuse_a=*/false, /*reuse_b=*/false);
  }
  const int n = bcol;
  const float bias = (n < 81) ? b_cls[n] : ((n < 85) ? b_box[n - 81] : 0.0f);
  const int mtop = (lane >> 4) * 8;            // C layout: lanes>=16 hold M+8
  for (int v = 0; v < 8; ++v)
    logits[(size_t)(mbase + mtop + v) * NCOLP + n] = acc[v] + bias;
}

// ------------------- 5. softmax + fg score + box decode --------------------
__global__ __launch_bounds__(256) void head_kernel(
    const float* __restrict__ logits, const float* __restrict__ props,
    float* __restrict__ boxes, float* __restrict__ scores) {
  const int idx = blockIdx.x * 256 + threadIdx.x;    // 0..511
  const float* l = logits + (size_t)idx * NCOLP;
  float mx = l[0];
  for (int i = 1; i < 81; ++i) mx = fmaxf(mx, l[i]);
  float sum = 0.0f, fg = 0.0f;
  for (int i = 0; i < 81; ++i) {
    const float e = __expf(l[i] - mx);
    sum += e;
    if (i < 80) fg = fmaxf(fg, e);
  }
  fg /= sum;
  const float* p = props + (size_t)idx * 4;
  const float w = p[2] - p[0], h = p[3] - p[1];
  const float cx = p[0] + 0.5f * w, cy = p[1] + 0.5f * h;
  const float dx = l[81] * 0.1f, dy = l[82] * 0.1f;           // /10
  const float dw = fminf(l[83] * 0.2f, SCALE_CLAMP);          // /5, clamp
  const float dh = fminf(l[84] * 0.2f, SCALE_CLAMP);
  const float pcx = dx * w + cx, pcy = dy * h + cy;
  const float pw = __expf(dw) * w, ph = __expf(dh) * h;
  float bx1 = pcx - 0.5f * pw, by1 = pcy - 0.5f * ph;
  float bx2 = pcx + 0.5f * pw, by2 = pcy + 0.5f * ph;
  bx1 = fminf(fmaxf(bx1, 0.0f), 1280.0f);
  by1 = fminf(fmaxf(by1, 0.0f),  800.0f);
  bx2 = fminf(fmaxf(bx2, 0.0f), 1280.0f);
  by2 = fminf(fmaxf(by2, 0.0f),  800.0f);
  float* b = boxes + (size_t)idx * 4;
  b[0] = bx1; b[1] = by1; b[2] = bx2; b[3] = by2;
  scores[idx] = fg;
}

// ------------------- 6. NMS + top-100 + det assembly -----------------------
__global__ __launch_bounds__(256) void nms_kernel(
    const float* __restrict__ boxes, const float* __restrict__ scores,
    float* __restrict__ out) {
  __shared__ float sb[NPROP][4];
  __shared__ float sArea[NPROP];
  __shared__ int   sOrder[NPROP];
  __shared__ int   sKeepSorted[NPROP];
  __shared__ int   keepOrig[NPROP];
  __shared__ float oScore[NPROP];
  __shared__ float masked[NPROP];
  const int n = blockIdx.x, t = threadIdx.x;
  const float* bsrc = boxes + (size_t)n * NPROP * 4;
  const float myScore = scores[n * NPROP + t];
  oScore[t] = myScore;
  __syncthreads();
  // stable descending rank (argsort(-scores))
  int r = 0;
  for (int j = 0; j < NPROP; ++j) {
    const float sj = oScore[j];
    if (sj > myScore || (sj == myScore && j < t)) ++r;
  }
  sOrder[r] = t;
  __syncthreads();
  const int oi = sOrder[t];
  const float4 bb = *(const float4*)(bsrc + oi * 4);
  sb[t][0] = bb.x; sb[t][1] = bb.y; sb[t][2] = bb.z; sb[t][3] = bb.w;
  sArea[t] = (bb.z - bb.x) * (bb.w - bb.y);
  sKeepSorted[t] = 1;
  __syncthreads();
  // sequential greedy NMS over sorted order; threads parallelize the j<k scan
  for (int k = 1; k < NPROP; ++k) {
    int pred = 0;
    if (t < k && sKeepSorted[t]) {
      const float ix1 = fmaxf(sb[t][0], sb[k][0]);
      const float iy1 = fmaxf(sb[t][1], sb[k][1]);
      const float ix2 = fminf(sb[t][2], sb[k][2]);
      const float iy2 = fminf(sb[t][3], sb[k][3]);
      const float iw = fmaxf(ix2 - ix1, 0.0f), ih = fmaxf(iy2 - iy1, 0.0f);
      const float inter = iw * ih;
      const float iou = inter / fmaxf(sArea[t] + sArea[k] - inter, 1e-9f);
      pred = (iou > 0.5f) ? 1 : 0;
    }
    const int sup = __syncthreads_or(pred);
    if (t == 0) sKeepSorted[k] = sup ? 0 : 1;
    __syncthreads();
  }
  keepOrig[sOrder[t]] = sKeepSorted[t];
  __syncthreads();
  int keep = keepOrig[t] && (myScore > 0.05f);
  masked[t] = keep ? myScore : -INFINITY;
  __syncthreads();
  const float mm = masked[t];
  int r2 = 0;
  for (int j = 0; j < NPROP; ++j) {
    const float mj = masked[j];
    if (mj > mm || (mj == mm && j < t)) ++r2;
  }
  keep = keep && (r2 < 100);
  const float4 myb = *(const float4*)(bsrc + t * 4);
  float* o = out + ((size_t)n * NPROP + t) * 6;
  o[0] = myb.x; o[1] = myb.y; o[2] = myb.z; o[3] = myb.w;
  o[4] = myScore; o[5] = keep ? 1.0f : 0.0f;
}

// ---------------------------------------------------------------------------
extern "C" void kernel_launch(void* const* d_in, const int* in_sizes, int n_in,
                              void* d_out, int out_size, void* d_ws, size_t ws_size,
                              hipStream_t stream) {
  (void)in_sizes; (void)n_in; (void)out_size; (void)ws_size;
  const float* features  = (const float*)d_in[0];   // (2,50,80,1024)
  const float* proposals = (const float*)d_in[1];   // (2,256,4)
  const float* gt_boxes  = (const float*)d_in[2];   // (2,64,4)
  // d_in[3] gt_classes: unused by the reference outputs
  const float* W_cls = (const float*)d_in[4];       // (1024,81)
  const float* b_cls = (const float*)d_in[5];       // (81)
  const float* W_box = (const float*)d_in[6];       // (1024,4)
  const float* b_box = (const float*)d_in[7];       // (4)
  float* out = (float*)d_out;                       // 4096 f32

  char* ws = (char*)d_ws;
  _Float16* feat16 = (_Float16*)ws;                               // 512*1024*2  = 1048576 B
  _Float16* WT     = (_Float16*)(ws + 1048576);                   // 96*1024*2   =  196608 B
  float*    logits = (float*)(ws + 1048576 + 196608);             // 512*96*4    =  196608 B
  float*    dboxes = (float*)(ws + 1048576 + 2 * 196608);         // 512*4*4     =    8192 B
  float*    dscore = (float*)(ws + 1048576 + 2 * 196608 + 8192);  // 512*4       =    2048 B

  matcher_kernel     <<<2, 256, 0, stream>>>(gt_boxes, proposals, out);
  prep_weights_kernel<<<(NCOLP * FEAT_C + 255) / 256, 256, 0, stream>>>(W_cls, W_box, WT);
  roi_align_kernel   <<<512, 256, 0, stream>>>(features, proposals, feat16);
  gemm_wmma_kernel   <<<32, 192, 0, stream>>>(feat16, WT, b_cls, b_box, logits);
  head_kernel        <<<2, 256, 0, stream>>>(logits, proposals, dboxes, dscore);
  nms_kernel         <<<2, 256, 0, stream>>>(dboxes, dscore, out);
}
